// TernaryEqProp_46084999086184
// MI455X (gfx1250) — matrix operand order repaired and checked by reading, not compiled, not run on hardware
//
#include <hip/hip_runtime.h>
#include <hip/hip_bf16.h>
#include <math.h>

// ---------------------------------------------------------------------------
// Ternary EqProp RNN on MI455X (gfx1250, wave32, WMMA f16 -> f32 accum)
//   x_proj = x @ tern(W_in)^T + b_in
//   repeat 30: h = tanh(x_proj + h @ tern(W_rec)^T + b_rec)
//   out = h @ tern(W_out)^T + b_out
// GEMMs are TN (A=[M,K], W=[N,K], both K-major) -> contiguous 16B tile rows.
// Global->LDS uses CDNA5 async-copy (global_load_async_to_lds_b128, ASYNCcnt)
// so the wave's steady-state work is ds_load_b128 -> v_wmma only.
// ---------------------------------------------------------------------------

typedef __attribute__((ext_vector_type(16))) _Float16 v16h;
typedef __attribute__((ext_vector_type(8)))  _Float16 v8h;
typedef __attribute__((ext_vector_type(4)))  _Float16 v4h;
typedef __attribute__((ext_vector_type(8)))  float    v8f;

#define BM   128
#define BN   128
#define BK   32
#define LDT  40      // padded LDS row stride in halves (80 B, 16B-aligned rows)
#define NTHREADS 256 // 8 wave32s

// ---------------------------------------------------------------------------
// CDNA5 async global->LDS copy (tracked with ASYNCcnt)
// ---------------------------------------------------------------------------
__device__ __forceinline__ void async_copy_b128(unsigned lds_off, const void* gaddr) {
    // VDST = LDS byte address VGPR, VADDR = 64-bit global address (GV mode)
    asm volatile("global_load_async_to_lds_b128 %0, %1, off"
                 :: "v"(lds_off), "v"(gaddr)
                 : "memory");
}
__device__ __forceinline__ void wait_async0() {
    asm volatile("s_wait_asynccnt 0x0" ::: "memory");
}

// low 32 bits of the flat shared address == LDS byte offset (ISA 10.2)
__device__ __forceinline__ unsigned lds_off(const void* p) {
    return (unsigned)(uintptr_t)p;
}

// ---------------------------------------------------------------------------
// Elementwise helpers
// ---------------------------------------------------------------------------
__device__ __forceinline__ float tern1(float w) {
    return (w > 0.5f) ? 1.0f : ((w < -0.5f) ? -1.0f : 0.0f);
}

// tanh from HW transcendentals: v_exp_f32 + v_rcp_f32 (plenty for f16 h)
__device__ __forceinline__ float fast_tanh(float x) {
    float ax = fabsf(x);
    float e  = __expf(-2.0f * ax);
    float t  = (1.0f - e) * __builtin_amdgcn_rcpf(1.0f + e);
    return copysignf(t, x);
}

// f32 -> ternary f16 ({-1,0,+1} exact in f16), 4 elems/thread
__global__ __launch_bounds__(256) void quantize_tern_f16(
    const float* __restrict__ w, _Float16* __restrict__ q, int n4) {
    int i = blockIdx.x * blockDim.x + threadIdx.x;
    if (i >= n4) return;
    const float4 v = *(const float4*)(w + (size_t)i * 4);
    v4h o;
    o[0] = (_Float16)tern1(v.x);
    o[1] = (_Float16)tern1(v.y);
    o[2] = (_Float16)tern1(v.z);
    o[3] = (_Float16)tern1(v.w);
    *(v4h*)(q + (size_t)i * 4) = o;
}

// f32 -> f16, 4 elems/thread
__global__ __launch_bounds__(256) void cvt_f32_f16(
    const float* __restrict__ s, _Float16* __restrict__ d, int n4) {
    int i = blockIdx.x * blockDim.x + threadIdx.x;
    if (i >= n4) return;
    const float4 v = *(const float4*)(s + (size_t)i * 4);
    v4h o;
    o[0] = (_Float16)v.x; o[1] = (_Float16)v.y;
    o[2] = (_Float16)v.z; o[3] = (_Float16)v.w;
    *(v4h*)(d + (size_t)i * 4) = o;
}

// zero-fill f16 buffer, 8 elems/thread
__global__ __launch_bounds__(256) void zero_f16(_Float16* __restrict__ d, int n8) {
    int i = blockIdx.x * blockDim.x + threadIdx.x;
    if (i >= n8) return;
    uint4 z; z.x = 0u; z.y = 0u; z.z = 0u; z.w = 0u;
    *(uint4*)(d + (size_t)i * 8) = z;
}

// ---------------------------------------------------------------------------
// Tiled WMMA GEMM:  out = epilogue( A[M,K] * Bw[N,K]^T + bias (+ addend) )
//   doTanh==1 : out_f16 = tanh(...)   (recurrent step, h for next iteration)
//   doTanh==0 : out_f32 = ...         (x_proj and final projection)
// 256 threads = 8 waves; wave grid 4(M) x 2(N); each wave -> 32x64 via
// 2x4 accumulators of v_wmma_f32_16x16x32_f16. Double-buffered LDS filled
// by async copies; one barrier + one s_wait_asynccnt per K-tile.
// ---------------------------------------------------------------------------
__global__ __launch_bounds__(NTHREADS) void gemm_wmma_f16(
    const _Float16* __restrict__ A,    // [M,K] f16 activations
    const _Float16* __restrict__ Bw,   // [N,K] f16 ternary weights
    const float*    __restrict__ bias, // [N]
    const float*    __restrict__ addend, // [M,N] f32 or nullptr
    float*          __restrict__ outF, // [M,N] f32 (doTanh==0)
    _Float16*       __restrict__ outH, // [M,N] f16 (doTanh==1)
    int M, int N, int K, int doTanh)
{
    __shared__ _Float16 sA[2][BM * LDT];
    __shared__ _Float16 sB[2][BN * LDT];

    const int tid  = threadIdx.x;
    const int lane = tid & 31;
    const int wave = tid >> 5;
    const int wm   = wave & 3;   // M macro-tile: wm*32
    const int wn   = wave >> 2;  // N macro-tile: wn*64
    const int r16  = lane & 15;
    const int kh   = lane >> 4;  // lane-half selector (ISA 7.12.2 layouts)

    const int bm = blockIdx.y * BM;
    const int bn = blockIdx.x * BN;

    // --- async global->LDS: 512 x 16B chunks per 128x32 tile, 2/thread ---
    const int c0  = tid, c1 = tid + NTHREADS;
    const int ar0 = c0 >> 2, ac0 = (c0 & 3) * 8;
    const int ar1 = c1 >> 2, ac1 = (c1 & 3) * 8;

    const _Float16* aBase = A  + (size_t)bm * K;
    const _Float16* bBase = Bw + (size_t)bn * K;

    // per-thread LDS destinations (both buffers), computed once
    unsigned dA0[2], dA1[2], dB0[2], dB1[2];
#pragma unroll
    for (int b = 0; b < 2; ++b) {
        dA0[b] = lds_off(&sA[b][ar0 * LDT + ac0]);
        dA1[b] = lds_off(&sA[b][ar1 * LDT + ac1]);
        dB0[b] = lds_off(&sB[b][ar0 * LDT + ac0]);
        dB1[b] = lds_off(&sB[b][ar1 * LDT + ac1]);
    }

    auto issueTile = [&](int kt, int b) {
        const size_t kofs = (size_t)kt * BK;
        async_copy_b128(dA0[b], aBase + (size_t)ar0 * K + kofs + ac0);
        async_copy_b128(dA1[b], aBase + (size_t)ar1 * K + kofs + ac1);
        async_copy_b128(dB0[b], bBase + (size_t)ar0 * K + kofs + ac0);
        async_copy_b128(dB1[b], bBase + (size_t)ar1 * K + kofs + ac1);
    };

    // --- fragment loads (ISA 7.12.2 VGPR layouts, wave32) ---
    // A 16x32: lane<16 holds K {0..7,16..23}; lane>=16 holds K {8..15,24..31}
    auto loadAfrag = [&](int b, int tm) -> v16h {
        const _Float16* p = &sA[b][(wm * 32 + tm * 16 + r16) * LDT];
        const v8h lo = *(const v8h*)(p + kh * 8);
        const v8h hi = *(const v8h*)(p + 16 + kh * 8);
        v16h a;
#pragma unroll
        for (int i = 0; i < 8; ++i) { a[i] = lo[i]; a[8 + i] = hi[i]; }
        return a;
    };
    // B 32x16: lane = N, lane<16 holds K {0..7,16..23}; lane>=16 holds K {8..15,24..31}
    auto loadBfrag = [&](int b, int tn) -> v16h {
        const _Float16* p = &sB[b][(wn * 64 + tn * 16 + r16) * LDT];
        const v8h lo = *(const v8h*)(p + kh * 8);
        const v8h hi = *(const v8h*)(p + 16 + kh * 8);
        v16h v;
#pragma unroll
        for (int i = 0; i < 8; ++i) { v[i] = lo[i]; v[8 + i] = hi[i]; }
        return v;
    };

    v8f acc[2][4];
#pragma unroll
    for (int tm = 0; tm < 2; ++tm)
#pragma unroll
        for (int tn = 0; tn < 4; ++tn)
            acc[tm][tn] = (v8f)(0.0f);

    const int KT = K / BK;

    // prologue: fill buffer 0
    issueTile(0, 0);
    wait_async0();
    __syncthreads();

    int buf = 0;
    for (int kt = 0; kt < KT; ++kt) {
        if (kt + 1 < KT) issueTile(kt + 1, buf ^ 1);   // async fill next buffer
        if (kt + 2 < KT) {                             // warm L2 two ahead
            const size_t pk = (size_t)(kt + 2) * BK;
            __builtin_prefetch(aBase + (size_t)ar0 * K + pk + ac0, 0, 1);
            __builtin_prefetch(bBase + (size_t)ar0 * K + pk + ac0, 0, 1);
        }

        v16h af[2], bf[4];
#pragma unroll
        for (int tm = 0; tm < 2; ++tm) af[tm] = loadAfrag(buf, tm);
#pragma unroll
        for (int tn = 0; tn < 4; ++tn) bf[tn] = loadBfrag(buf, tn);

#pragma unroll
        for (int tm = 0; tm < 2; ++tm)
#pragma unroll
            for (int tn = 0; tn < 4; ++tn)
                acc[tm][tn] = __builtin_amdgcn_wmma_f32_16x16x32_f16(
                    /*neg_a=*/false, af[tm], /*neg_b=*/false, bf[tn],
                    /*c_mod=*/(short)0, acc[tm][tn],
                    /*reuse_a=*/false, /*reuse_b=*/false);

        wait_async0();      // our async fills for kt+1 complete
        __syncthreads();    // everyone's fills visible; reads of buf done
        buf ^= 1;
    }

    // --- epilogue: C layout = VGPR r -> M = 8*kh + r, N = r16 (per tile) ---
    const int mBase = bm + wm * 32;
    const int nBase = bn + wn * 64;
#pragma unroll
    for (int tm = 0; tm < 2; ++tm) {
#pragma unroll
        for (int tn = 0; tn < 4; ++tn) {
            const int n  = nBase + tn * 16 + r16;
            const float bv = bias[n];
#pragma unroll
            for (int r = 0; r < 8; ++r) {
                const int m = mBase + tm * 16 + kh * 8 + r;
                const size_t idx = (size_t)m * N + n;
                float v = acc[tm][tn][r] + bv;
                if (addend) v += addend[idx];
                if (doTanh) outH[idx] = (_Float16)fast_tanh(v);
                else        outF[idx] = v;
            }
        }
    }
}

// ---------------------------------------------------------------------------
// Launch: quantize weights once, project, 30 recurrent WMMA steps, project.
// ---------------------------------------------------------------------------
extern "C" void kernel_launch(void* const* d_in, const int* in_sizes, int n_in,
                              void* d_out, int out_size, void* d_ws, size_t ws_size,
                              hipStream_t stream) {
    (void)in_sizes; (void)n_in; (void)out_size; (void)ws_size;

    const float* x     = (const float*)d_in[0];
    const float* W_in  = (const float*)d_in[1];
    const float* b_in  = (const float*)d_in[2];
    const float* W_rec = (const float*)d_in[3];
    const float* b_rec = (const float*)d_in[4];
    const float* W_out = (const float*)d_in[5];
    const float* b_out = (const float*)d_in[6];
    float* out = (float*)d_out;

    const int Bsz = 1024, Din = 2048, Dh = 4096, Dout = 1024;
    const int MAX_STEPS = 30;

    // carve workspace (total ~92 MiB)
    char* p = (char*)d_ws;
    auto carve = [&](size_t bytes) -> char* {
        char* r = p;
        p += (bytes + 255) & ~(size_t)255;
        return r;
    };
    _Float16* Qin   = (_Float16*)carve((size_t)Dh  * Din * 2);
    _Float16* Qrec  = (_Float16*)carve((size_t)Dh  * Dh  * 2);
    _Float16* Qout  = (_Float16*)carve((size_t)Dout* Dh  * 2);
    _Float16* xh    = (_Float16*)carve((size_t)Bsz * Din * 2);
    float*    xproj = (float*)   carve((size_t)Bsz * Dh  * 4);
    _Float16* h0    = (_Float16*)carve((size_t)Bsz * Dh  * 2);
    _Float16* h1    = (_Float16*)carve((size_t)Bsz * Dh  * 2);

    const int T = 256;
    auto blocks4 = [](size_t n) { return (unsigned)((n / 4 + 255) / 256); };

    // one-time precision conversion (bandwidth-bound, L2-resident afterwards)
    quantize_tern_f16<<<blocks4((size_t)Dh * Din), T, 0, stream>>>(W_in,  Qin,  Dh * Din / 4);
    quantize_tern_f16<<<blocks4((size_t)Dh * Dh ), T, 0, stream>>>(W_rec, Qrec, Dh * Dh  / 4);
    quantize_tern_f16<<<blocks4((size_t)Dout* Dh ), T, 0, stream>>>(W_out, Qout, Dout * Dh / 4);
    cvt_f32_f16<<<blocks4((size_t)Bsz * Din), T, 0, stream>>>(x, xh, Bsz * Din / 4);
    zero_f16<<<(unsigned)((size_t)Bsz * Dh / 8 / 256), T, 0, stream>>>(h0, Bsz * Dh / 8);

    // x_proj = x @ Qin^T + b_in                        [1024 x 4096, K=2048]
    {
        dim3 g(Dh / BN, Bsz / BM);
        gemm_wmma_f16<<<g, NTHREADS, 0, stream>>>(
            xh, Qin, b_in, nullptr, xproj, nullptr, Bsz, Dh, Din, 0);
    }

    // 30 x: h = tanh(x_proj + h @ Qrec^T + b_rec)      [1024 x 4096, K=4096]
    {
        dim3 g(Dh / BN, Bsz / BM);
        for (int i = 0; i < MAX_STEPS; ++i) {
            const _Float16* hin  = (i & 1) ? h1 : h0;
            _Float16*       hout = (i & 1) ? h0 : h1;
            gemm_wmma_f16<<<g, NTHREADS, 0, stream>>>(
                hin, Qrec, b_rec, xproj, nullptr, hout, Bsz, Dh, Dh, 1);
        }
    }

    // out = h @ Qout^T + b_out                         [1024 x 1024, K=4096]
    {
        dim3 g(Dout / BN, Bsz / BM);
        gemm_wmma_f16<<<g, NTHREADS, 0, stream>>>(
            h0, Qout, b_out, nullptr, out, nullptr, Bsz, Dout, Dh, 0);
    }
}
